// GCN_new_61512521613334
// MI455X (gfx1250) — matrix operand order repaired and checked by reading, not compiled, not run on hardware
//
#include <hip/hip_runtime.h>
#include <hip/hip_bf16.h>

#define NNODES 150000
#define NEDGES 600000
#define INDIM  128
#define HID    64

typedef __attribute__((ext_vector_type(16))) _Float16 v16h;
typedef __attribute__((ext_vector_type(8)))  float    v8f;

// ---------------------------------------------------------------------------
// Degree / normalization kernels
// ---------------------------------------------------------------------------
__global__ void deg_init_kernel(float* __restrict__ deg, int n) {
    int i = blockIdx.x * blockDim.x + threadIdx.x;
    if (i < n) deg[i] = 1.0f;            // self-loop contributes 1
}

__global__ void deg_acc_kernel(const int* __restrict__ col, float* __restrict__ deg, int e) {
    int i = blockIdx.x * blockDim.x + threadIdx.x;
    if (i < e) atomicAdd(&deg[col[i]], 1.0f);
}

__global__ void dinv_kernel(float* __restrict__ deg_dinv, int n) {
    int i = blockIdx.x * blockDim.x + threadIdx.x;
    if (i < n) deg_dinv[i] = rsqrtf(deg_dinv[i]);   // deg >= 1 always
}

// ---------------------------------------------------------------------------
// WMMA GEMM:  Y[n][HID] = X[n][K] @ W^T   (W is [HID][K] row-major, f32)
// One wave -> 16 rows x 64 cols.  f16 inputs, f32 accumulate.
// ---------------------------------------------------------------------------
template <int K>
__global__ __launch_bounds__(256) void gemm_wmma_kernel(
    const float* __restrict__ X, const float* __restrict__ Wm,
    float* __restrict__ Y, int n)
{
    __shared__ __align__(32) _Float16 ldsW[HID * K];

    const int tid = threadIdx.x;
    for (int i = tid; i < HID * K; i += 256) ldsW[i] = (_Float16)Wm[i];
    __syncthreads();

    const int wave = tid >> 5;
    const int lane = tid & 31;
    const int g    = lane >> 4;   // half-wave group
    const int m    = lane & 15;   // row-in-tile (A) / col-in-tile (B,C)

    const long long rowbase = (long long)(blockIdx.x * 8 + wave) * 16;

    long long arow = rowbase + m;                 // A-source row, clamped so
    if (arow > (long long)n - 1) arow = n - 1;    // EXEC stays all-ones
    const float* __restrict__ xrow = X + arow * K;

    v8f acc[4];
    #pragma unroll
    for (int t = 0; t < 4; ++t) acc[t] = v8f{};

    #pragma unroll
    for (int kb = 0; kb < K; kb += 32) {
        // A fragment: half h -> k = kb+16g+h
        v16h a;
        const float* p0 = xrow + kb + 16 * g;
        const float* p1 = xrow + kb + 16 * g + 8;
        #pragma unroll
        for (int h = 0; h < 8; ++h) a[h]     = (_Float16)p0[h];
        #pragma unroll
        for (int h = 0; h < 8; ++h) a[8 + h] = (_Float16)p1[h];

        // B fragments: lane holds column (t*16+m), k = kb + 16g .. +15
        #pragma unroll
        for (int t = 0; t < 4; ++t) {
            const v16h b = *(const v16h*)&ldsW[(t * 16 + m) * K + kb + 16 * g];
            acc[t] = __builtin_amdgcn_wmma_f32_16x16x32_f16(
                false, a, false, b, (short)0, acc[t], false, false);
        }
    }

    // C layout: vgpr r, lane-group g -> row M = r + 8g ; col N = t*16 + m
    #pragma unroll
    for (int t = 0; t < 4; ++t) {
        #pragma unroll
        for (int r = 0; r < 8; ++r) {
            const long long rr = rowbase + 8 * g + r;
            if (rr < n) Y[rr * HID + t * 16 + m] = acc[t][r];
        }
    }
}

// ---------------------------------------------------------------------------
// Zero, scatter, epilogue, head
// ---------------------------------------------------------------------------
__global__ void zero_kernel(float4* __restrict__ p, int n4) {
    int i = blockIdx.x * blockDim.x + threadIdx.x;
    if (i < n4) p[i] = make_float4(0.f, 0.f, 0.f, 0.f);
}

// out[col] += dinv[row]*dinv[col] * xw[row]  (64 feats per edge, 1 thread/feat)
__global__ void scatter_kernel(const float* __restrict__ xw,
                               const int* __restrict__ row,
                               const int* __restrict__ col,
                               const float* __restrict__ dinv,
                               float* __restrict__ acc, int e)
{
    long long idx = (long long)blockIdx.x * blockDim.x + threadIdx.x;
    int ei = (int)(idx >> 6);
    int f  = (int)(idx & 63);
    if (ei >= e) return;
    const int r = row[ei], c = col[ei];
    const float nrm = dinv[r] * dinv[c];
    atomicAdd(&acc[(long long)c * HID + f], nrm * xw[(long long)r * HID + f]);
}

// h = relu(acc + dinv^2 * xw + bias)   (in-place on acc is safe)
__global__ void finish_kernel(const float* __restrict__ acc,
                              const float* __restrict__ xw,
                              const float* __restrict__ dinv,
                              const float* __restrict__ bias,
                              float* __restrict__ h, int n)
{
    long long idx = (long long)blockIdx.x * blockDim.x + threadIdx.x;
    int i = (int)(idx >> 6);
    int f = (int)(idx & 63);
    if (i >= n) return;
    const float s = dinv[i];
    const float v = acc[idx] + s * s * xw[idx] + bias[f];
    h[idx] = v > 0.f ? v : 0.f;
}

// out[j] = b3 + dot(h2[grp*15+3+k], W3);  j = grp*12 + k
__global__ void head_kernel(const float* __restrict__ h2,
                            const float* __restrict__ W3,
                            const float* __restrict__ b3,
                            float* __restrict__ out, int nout)
{
    int j = blockIdx.x * blockDim.x + threadIdx.x;
    if (j >= nout) return;
    const int grp = j / 12;
    const int k   = j % 12;
    const long long i = (long long)grp * 15 + 3 + k;
    const float* __restrict__ hr = h2 + i * HID;
    float s = b3[0];
    #pragma unroll
    for (int f = 0; f < HID; ++f) s += hr[f] * W3[f];
    out[j] = s;
}

// ---------------------------------------------------------------------------
// Launch
// ---------------------------------------------------------------------------
extern "C" void kernel_launch(void* const* d_in, const int* in_sizes, int n_in,
                              void* d_out, int out_size, void* d_ws, size_t ws_size,
                              hipStream_t stream)
{
    const float* obs = (const float*)d_in[0];
    const int*   ei  = (const int*)d_in[1];       // [2][E]: row then col
    const float* W1  = (const float*)d_in[2];
    const float* b1  = (const float*)d_in[3];
    const float* W2  = (const float*)d_in[4];
    const float* b2  = (const float*)d_in[5];
    const float* W3  = (const float*)d_in[6];
    const float* b3  = (const float*)d_in[7];
    float* out = (float*)d_out;

    const int N = NNODES, E = NEDGES;
    const int* row = ei;
    const int* col = ei + E;

    // workspace layout (floats)
    float* ws   = (float*)d_ws;
    float* dinv = ws;                                  // N (padded to 150016)
    float* bufA = ws + 150016;                         // N*HID : xw
    float* bufB = bufA + (size_t)N * HID;              // N*HID : acc1 -> h1
    float* bufC = bufB + (size_t)N * HID;              // N*HID : acc2 -> h2

    const int NH   = N * HID;                          // 9,600,000
    const int NH4  = NH / 4;                           // 2,400,000
    const int TPB  = 256;

    // 1) symmetric normalization
    deg_init_kernel<<<(N + TPB - 1) / TPB, TPB, 0, stream>>>(dinv, N);
    deg_acc_kernel <<<(E + TPB - 1) / TPB, TPB, 0, stream>>>(col, dinv, E);
    dinv_kernel    <<<(N + TPB - 1) / TPB, TPB, 0, stream>>>(dinv, N);

    const int gemm_blocks = (N + 127) / 128;           // 8 waves x 16 rows

    // 2) layer 1: xw = obs @ W1^T ; acc = scatter ; h1 = relu(acc+self+b1)
    gemm_wmma_kernel<INDIM><<<gemm_blocks, TPB, 0, stream>>>(obs, W1, bufA, N);
    zero_kernel<<<(NH4 + TPB - 1) / TPB, TPB, 0, stream>>>((float4*)bufB, NH4);
    {
        long long th = (long long)E * HID;
        scatter_kernel<<<(int)((th + TPB - 1) / TPB), TPB, 0, stream>>>(
            bufA, row, col, dinv, bufB, E);
    }
    finish_kernel<<<(NH + TPB - 1) / TPB, TPB, 0, stream>>>(bufB, bufA, dinv, b1, bufB, N);

    // 3) layer 2: xw = h1 @ W2^T ; acc = scatter ; h2 = relu(acc+self+b2)
    gemm_wmma_kernel<HID><<<gemm_blocks, TPB, 0, stream>>>(bufB, W2, bufA, N);
    zero_kernel<<<(NH4 + TPB - 1) / TPB, TPB, 0, stream>>>((float4*)bufC, NH4);
    {
        long long th = (long long)E * HID;
        scatter_kernel<<<(int)((th + TPB - 1) / TPB), TPB, 0, stream>>>(
            bufA, row, col, dinv, bufC, E);
    }
    finish_kernel<<<(NH + TPB - 1) / TPB, TPB, 0, stream>>>(bufC, bufA, dinv, b2, bufC, N);

    // 4) head + reshape-drop
    head_kernel<<<(out_size + TPB - 1) / TPB, TPB, 0, stream>>>(bufC, W3, b3, out, out_size);
}
